// FunctionLibrary_89696097010332
// MI455X (gfx1250) — matrix operand order repaired and checked by reading, not compiled, not run on hardware
//
#include <hip/hip_runtime.h>
#include <hip/hip_bf16.h>
#include <math.h>

// ---------------------------------------------------------------------------
// Problem constants (from reference): N=16384, K=32, D=512, F=8
// ---------------------------------------------------------------------------
#define PN 16384
#define PK 32
#define PD 512
#define PF 8
#define TWO_D 1024

typedef __attribute__((ext_vector_type(16))) __bf16 v16bf;
typedef __attribute__((ext_vector_type(8)))  float  v8f;

__device__ __forceinline__ unsigned int pack_bf16(float a, float b) {
    unsigned int ua = __float_as_uint(a), ub = __float_as_uint(b);
    ua = (ua + 0x7FFFu + ((ua >> 16) & 1u)) >> 16;
    ub = (ub + 0x7FFFu + ((ub >> 16) & 1u)) >> 16;
    return ua | (ub << 16);
}

// ---------------------------------------------------------------------------
// Kernel 1: convert net_W / gate_W (f32 [F,2D,D]) to bf16, pre-swizzled into
// WMMA B-fragment order:
//   frag index = (((f*32 + kk)*32 + ct)*32 + lane), each frag = 16 bf16 (32B)
//   lane<16 : col = ct*16 + lane,    K = kk*32 + 0..15   (2 per dword)
//   lane>=16: col = ct*16 + lane-16, K = kk*32 + 16..31
// One thread produces one packed dword (2 bf16).
// ---------------------------------------------------------------------------
__global__ void k_convert_weights(const float* __restrict__ netW,
                                  const float* __restrict__ gateW,
                                  unsigned int* __restrict__ outNet,
                                  unsigned int* __restrict__ outGate) {
    const long per = (long)PF * 32 * 32 * 32 * 8;   // 2,097,152 dwords per tensor
    long id = (long)blockIdx.x * 256 + threadIdx.x;
    if (id >= 2 * per) return;
    const float* src = netW;
    unsigned int* dst = outNet;
    long lid = id;
    if (lid >= per) { lid -= per; src = gateW; dst = outGate; }

    int v    = (int)(lid & 7);        long u = lid >> 3;
    int lane = (int)(u & 31);         u >>= 5;
    int ct   = (int)(u & 31);         u >>= 5;
    int kk   = (int)(u & 31);         u >>= 5;
    int f    = (int)u;

    int col = ct * 16 + (lane & 15);
    int k0  = kk * 32 + ((lane & 16) ? 16 : 0) + 2 * v;
    const float* base = src + ((long)f * TWO_D + k0) * PD + col;
    dst[lid] = pack_bf16(base[0], base[PD]);   // base[PD] = next K row
}

// ---------------------------------------------------------------------------
// Kernel 2: per-row prep (bandwidth bound: streams routed_tokens, 1.07 GB)
//   - fn_w  = softmax(token @ sel_W + sel_b)        -> ws (f32 [N,8])
//   - arg_w = softmax(sim_row); argument = arg_w @ routed_tokens
//   - inp_bf16[n] = bf16([token ; argument])        -> ws ([N,512] dwords)
// One block of 256 threads per row n.
// ---------------------------------------------------------------------------
__global__ void k_prep(const float* __restrict__ token,
                       const float* __restrict__ routed,
                       const float* __restrict__ sim,
                       const float* __restrict__ selW,
                       const float* __restrict__ selb,
                       unsigned int* __restrict__ inp,
                       float* __restrict__ fnw) {
    __shared__ float sRed[64];
    __shared__ float sArg[32];
    __shared__ float sFw[8];
    const int t = threadIdx.x, lane = t & 31, wave = t >> 5;
    const long n = blockIdx.x;
    const float* tok = token + n * PD;

    // selector logits: partial dot(token, sel_W[:,f])
    float p[PF];
#pragma unroll
    for (int f = 0; f < PF; ++f) p[f] = 0.f;
    for (int d = t; d < PD; d += 256) {
        float x = tok[d];
        const float* wrow = selW + d * PF;
#pragma unroll
        for (int f = 0; f < PF; ++f) p[f] += x * wrow[f];
    }
#pragma unroll
    for (int f = 0; f < PF; ++f) {
#pragma unroll
        for (int m = 16; m >= 1; m >>= 1) p[f] += __shfl_xor(p[f], m, 32);
    }
    if (lane == 0) {
#pragma unroll
        for (int f = 0; f < PF; ++f) sRed[wave * PF + f] = p[f];
    }
    // arg_w softmax over K=32 on wave 0
    if (wave == 0) {
        float x = sim[n * PK + lane];
        float mx = x;
#pragma unroll
        for (int m = 16; m >= 1; m >>= 1) mx = fmaxf(mx, __shfl_xor(mx, m, 32));
        float e = expf(x - mx);
        float s = e;
#pragma unroll
        for (int m = 16; m >= 1; m >>= 1) s += __shfl_xor(s, m, 32);
        sArg[lane] = e / s;
    }
    __syncthreads();
    if (t < PF) {
        float s = selb[t];
#pragma unroll
        for (int w = 0; w < 8; ++w) s += sRed[w * PF + t];
        sFw[t] = s;
    }
    __syncthreads();
    if (t == 0) {
        float mx = sFw[0];
        for (int f = 1; f < PF; ++f) mx = fmaxf(mx, sFw[f]);
        float e[PF], s = 0.f;
        for (int f = 0; f < PF; ++f) { e[f] = expf(sFw[f] - mx); s += e[f]; }
        float inv = 1.f / s;
        for (int f = 0; f < PF; ++f) fnw[n * PF + f] = e[f] * inv;
    }
    __syncthreads();

    // argument = sum_k arg_w[k] * routed[n,k,:]; each thread owns 2 columns
    const float2* R2 = (const float2*)(routed + n * PK * PD);
    float a0 = 0.f, a1 = 0.f;
    for (int k = 0; k < PK; ++k) {
        float w = sArg[k];
        float2 r = R2[k * (PD / 2) + t];
        a0 += w * r.x;
        a1 += w * r.y;
    }
    const float2 tk = ((const float2*)tok)[t];
    unsigned int* row = inp + n * (TWO_D / 2);
    row[t]       = pack_bf16(tk.x, tk.y);  // first 512 bf16 = token
    row[256 + t] = pack_bf16(a0, a1);      // next 512 bf16 = argument
}

// ---------------------------------------------------------------------------
// Kernel 3: grouped WMMA GEMM + epilogue.
// Block = 16 rows x 256 threads (8 waves). Wave w owns columns [64w, 64w+64)
// as 4x (16x16) tiles. Loop over F=8 experts, K=1024 in 32 steps of
// v_wmma_f32_16x16x32_bf16 (net + gate simultaneously -> 8 WMMA / k-step).
// Epilogue: bias+GELU, cross-wave LayerNorm stats via LDS, sigmoid gate,
// softmax-weighted accumulation over experts in registers, single store.
// ---------------------------------------------------------------------------
__global__ __launch_bounds__(256) void k_gemm(
    const unsigned int* __restrict__ inp,   // [N][512] dwords (bf16x2)
    const float* __restrict__ fnw,          // [N][8]
    const unsigned int* __restrict__ BnU,   // swizzled net bf16 fragments
    const unsigned int* __restrict__ BgU,   // swizzled gate bf16 fragments
    const float* __restrict__ netb, const float* __restrict__ gateb,
    const float* __restrict__ lng,  const float* __restrict__ lnb,
    float* __restrict__ out, int N) {
    __shared__ __align__(16) unsigned int sA[16 * 516];  // 2064B row stride (bank pad)
    __shared__ float sS1[128], sS2[128], sMu[16], sRs[16], sFw[128];

    const int t = threadIdx.x, lane = t & 31, wave = t >> 5;
    const int lo = lane & 15;
    const int half8 = (lane & 16) ? 8 : 0;
    const long rowBase = (long)blockIdx.x * 16;

    // cooperative A-tile load: 16 rows x 2048B -> LDS (b128 chunks)
    {
        const uint4* g = (const uint4*)(inp + rowBase * (TWO_D / 2));
#pragma unroll
        for (int it = 0; it < 8; ++it) {
            int idx = it * 256 + t;          // uint4 units, 128 per row
            int r = idx >> 7, c = idx & 127;
            *(uint4*)&sA[r * 516 + c * 4] = g[r * 128 + c];
        }
    }
    if (t < 128) sFw[t] = fnw[rowBase * PF + t];
    __syncthreads();

    const v16bf* BnV = (const v16bf*)BnU;
    const v16bf* BgV = (const v16bf*)BgU;

    v8f outp[4] = {}, outg[4] = {};

    for (int f = 0; f < PF; ++f) {
        v8f ah[4] = {}, ag[4] = {};
        // fragment index: f*32768 + kk*1024 + (wave*4+ct)*32 + lane
        const long fb = (long)f * 32768 + (long)wave * 128 + lane;

        for (int kk = 0; kk < 32; ++kk) {
            // A fragment from LDS: 2x 16B reads per lane
            union { v16bf v; uint4 q[2]; } A;
            const unsigned int* ap = &sA[lo * 516 + kk * 16 + ((lane & 16) ? 8 : 0)];
            A.q[0] = *(const uint4*)(ap);
            A.q[1] = *(const uint4*)(ap + 4);

            const long kb2 = fb + (long)kk * 1024;
            __builtin_prefetch(BnV + kb2 + 1024, 0, 1);  // next k-step stream
            __builtin_prefetch(BgV + kb2 + 1024, 0, 1);
#pragma unroll
            for (int ct = 0; ct < 4; ++ct) {
                v16bf bn = BnV[kb2 + ct * 32];
                v16bf bg = BgV[kb2 + ct * 32];
                ah[ct] = __builtin_amdgcn_wmma_f32_16x16x32_bf16(
                    false, A.v, false, bn, (short)0, ah[ct], false, false);
                ag[ct] = __builtin_amdgcn_wmma_f32_16x16x32_bf16(
                    false, A.v, false, bg, (short)0, ag[ct], false, false);
            }
        }

        // ---- epilogue for expert f ----
        float bh[4], bgg[4], lg[4], lb[4];
#pragma unroll
        for (int ct = 0; ct < 4; ++ct) {
            int col = wave * 64 + ct * 16 + lo;
            bh[ct]  = netb[f * PD + col];
            bgg[ct] = gateb[f * PD + col];
            lg[ct]  = lng[f * PD + col];
            lb[ct]  = lnb[f * PD + col];
        }
        float hg[4][8];
        float s1[8], s2[8];
#pragma unroll
        for (int i = 0; i < 8; ++i) { s1[i] = 0.f; s2[i] = 0.f; }
#pragma unroll
        for (int ct = 0; ct < 4; ++ct) {
#pragma unroll
            for (int i = 0; i < 8; ++i) {
                float h = ah[ct][i] + bh[ct];
                float e = 0.5f * h * (1.0f + erff(h * 0.70710678118654752f));
                hg[ct][i] = e;
                s1[i] += e;
                s2[i] += e * e;
            }
        }
        // reduce over 16 lanes (columns within wave), stay inside half-wave
#pragma unroll
        for (int i = 0; i < 8; ++i) {
#pragma unroll
            for (int m = 1; m < 16; m <<= 1) {
                s1[i] += __shfl_xor(s1[i], m, 32);
                s2[i] += __shfl_xor(s2[i], m, 32);
            }
        }
        if (lo == 0) {
#pragma unroll
            for (int i = 0; i < 8; ++i) {
                sS1[(half8 + i) * 8 + wave] = s1[i];
                sS2[(half8 + i) * 8 + wave] = s2[i];
            }
        }
        __syncthreads();
        if (t < 16) {
            float a = 0.f, b = 0.f;
#pragma unroll
            for (int w = 0; w < 8; ++w) { a += sS1[t * 8 + w]; b += sS2[t * 8 + w]; }
            float mu  = a * (1.0f / 512.0f);
            float var = b * (1.0f / 512.0f) - mu * mu;
            sMu[t] = mu;
            sRs[t] = rsqrtf(var + 1e-5f);
        }
        __syncthreads();
#pragma unroll
        for (int ct = 0; ct < 4; ++ct) {
#pragma unroll
            for (int i = 0; i < 8; ++i) {
                int row = half8 + i;
                float w    = sFw[row * PF + f];
                float prop = (hg[ct][i] - sMu[row]) * sRs[row] * lg[ct] + lb[ct];
                float gv   = ag[ct][i] + bgg[ct];
                gv = 1.0f / (1.0f + expf(-gv));
                outp[ct][i] += w * prop;
                outg[ct][i] += w * gv;
            }
        }
        __syncthreads();  // before stats LDS reuse by next expert
    }

    // final store: out = [weighted_proposed (N*D) ; weighted_gate (N*D)]
    const long gOff = (long)N * PD;
#pragma unroll
    for (int ct = 0; ct < 4; ++ct) {
#pragma unroll
        for (int i = 0; i < 8; ++i) {
            long n = rowBase + half8 + i;
            int col = wave * 64 + ct * 16 + lo;
            out[n * PD + col] = outp[ct][i];
            out[gOff + n * PD + col] = outg[ct][i];
        }
    }
}

// ---------------------------------------------------------------------------
// Host launcher
// ---------------------------------------------------------------------------
extern "C" void kernel_launch(void* const* d_in, const int* in_sizes, int n_in,
                              void* d_out, int out_size, void* d_ws, size_t ws_size,
                              hipStream_t stream) {
    const float* token  = (const float*)d_in[0];
    const float* routed = (const float*)d_in[1];
    const float* sim    = (const float*)d_in[2];
    const float* selW   = (const float*)d_in[3];
    const float* selb   = (const float*)d_in[4];
    const float* netW   = (const float*)d_in[5];
    const float* netb   = (const float*)d_in[6];
    const float* lng    = (const float*)d_in[7];
    const float* lnb    = (const float*)d_in[8];
    const float* gateW  = (const float*)d_in[9];
    const float* gateb  = (const float*)d_in[10];
    float* out = (float*)d_out;

    (void)in_sizes; (void)n_in; (void)out_size; (void)ws_size;

    // workspace layout
    char* ws = (char*)d_ws;
    unsigned int* inp = (unsigned int*)ws;                       // N*512 dwords = 32 MB
    size_t off = (size_t)PN * (TWO_D / 2) * 4;
    float* fnw = (float*)(ws + off);                             // N*8 f32 = 512 KB
    off += (size_t)PN * PF * 4;
    unsigned int* WnU = (unsigned int*)(ws + off);               // 8.0 MB swizzled bf16
    const size_t wbytes = (size_t)PF * TWO_D * PD * 2;           // per tensor
    off += wbytes;
    unsigned int* WgU = (unsigned int*)(ws + off);
    off += wbytes;

    // 1) weight convert+swizzle: 2 * 2,097,152 dwords / 256 = 16384 blocks
    k_convert_weights<<<16384, 256, 0, stream>>>(netW, gateW, WnU, WgU);
    // 2) per-row prep (streams routed_tokens once)
    k_prep<<<PN, 256, 0, stream>>>(token, routed, sim, selW, selb, inp, fnw);
    // 3) WMMA grouped GEMM + fused epilogue
    k_gemm<<<PN / 16, 256, 0, stream>>>(inp, fnw, WnU, WgU,
                                        netb, gateb, lng, lnb, out, PN);
}
